// MambaVisionMixer_24704651887166
// MI455X (gfx1250) — matrix operand order repaired and checked by reading, not compiled, not run on hardware
//
#include <hip/hip_runtime.h>
#include <hip/hip_bf16.h>
#include <stdint.h>

// ---------------------------------------------------------------------------
// MambaVisionMixer on MI455X (gfx1250)
//   bf16 WMMA GEMMs, double-buffered TDM weight panels in LDS, fp32 scan
//   B=8, L=2048, D_MODEL=512, D_INNER=1024, D_HALF=512, N_STATE=16, DT_RANK=32
// ---------------------------------------------------------------------------

#define B_SZ      8
#define SEQ       2048
#define D_MODEL   512
#define D_INNER   1024
#define D_HALF    512
#define N_STATE   16
#define DT_RANK   32
#define M_ROWS    (B_SZ * SEQ)          // 16384

typedef __attribute__((ext_vector_type(16))) __bf16 v16bf;
typedef __attribute__((ext_vector_type(8)))  __bf16 v8bf;
typedef __attribute__((ext_vector_type(4)))  __bf16 v4bf;
typedef __attribute__((ext_vector_type(8)))  float  v8f;
typedef __attribute__((ext_vector_type(4)))  unsigned v4u;
typedef __attribute__((ext_vector_type(8)))  int v8i;
typedef __attribute__((ext_vector_type(4)))  int v4i;

#if defined(__has_builtin)
#if __has_builtin(__builtin_amdgcn_tensor_load_to_lds)
#define HAS_TDM 1
#endif
#endif
#ifndef HAS_TDM
#define HAS_TDM 0
#endif

__device__ inline __bf16 f2bf(float x) {           // RNE f32 -> bf16
    union { float f; unsigned u; } v; v.f = x;
    unsigned r = v.u + 0x7FFFu + ((v.u >> 16) & 1u);
    unsigned short h = (unsigned short)(r >> 16);
    __bf16 out; __builtin_memcpy(&out, &h, sizeof(out));
    return out;
}
__device__ inline float bf2f(__bf16 b) {
    unsigned short h; __builtin_memcpy(&h, &b, sizeof(h));
    unsigned u = (unsigned)h << 16;
    float f; __builtin_memcpy(&f, &u, sizeof(f));
    return f;
}
__device__ inline float softplusf(float x) {
    return (x > 20.f) ? x : log1pf(__expf(x));
}

// One-off f32 -> bf16 cast (n multiple of 4).
__global__ void __launch_bounds__(256)
cast_f32_bf16(const float* __restrict__ in, __bf16* __restrict__ out, int n) {
    int i = (blockIdx.x * blockDim.x + threadIdx.x) * 4;
    if (i >= n) return;
    float4 v = *(const float4*)(in + i);
    v4bf o; o[0] = f2bf(v.x); o[1] = f2bf(v.y); o[2] = f2bf(v.z); o[3] = f2bf(v.w);
    *(v4bf*)(out + i) = o;
}

// ---------------------------------------------------------------------------
// Panel GEMM: out[m,n] = sum_k A[m,k] * W[n,k]
//   A bf16 [M,K] (columns [0,ksplit) from A0, [ksplit,K) from A1;
//                 ksplit must be a multiple of KCHUNK)
//   W bf16 [N,K] row-major (ldw == KT)
// Block = 256 threads = 8 waves; block tile = 128(M) x 64(N); wave tile 16x64.
// 64 x KCHUNK weight panels double-buffered in LDS via the Tensor Data Mover.
// ---------------------------------------------------------------------------
template <int KT, bool SOFTPLUS, bool WF32, bool WBF>
__global__ void __launch_bounds__(256)
gemm_wmma_panel(const __bf16* __restrict__ A0, int lda0,
                const __bf16* __restrict__ A1, int lda1, int ksplit,
                const __bf16* __restrict__ W,
                const float* __restrict__ bias,
                float* __restrict__ outf, __bf16* __restrict__ outb, int ldo,
                int M, int N) {
    constexpr int KCHUNK  = (KT < 256) ? KT : 256;
    constexpr int NCHUNKS = KT / KCHUNK;
    __shared__ __bf16 ldsW[2][64 * KCHUNK];          // 2 x <=32 KB

    const int nPanels = N >> 6;
    const int bm = blockIdx.x / nPanels;
    const int n0 = (blockIdx.x % nPanels) << 6;

    const int wid  = threadIdx.x >> 5;               // wave id: M direction
    const int lane = threadIdx.x & 31;
    const int r    = lane & 15;
    const int half = lane >> 4;
    const int tm   = (bm << 7) + (wid << 4);
    const int arow = tm + r;

    // A-side pointer for a whole chunk (chunks never straddle ksplit).
    auto a_chunk_ptr = [&](int kc0) -> const __bf16* {
        if (kc0 < ksplit) return A0 + (size_t)arow * lda0 + kc0 + half * 8;
        return A1 + (size_t)arow * lda1 + (kc0 - ksplit) + half * 8;
    };

    v8f acc[4];
#pragma unroll
    for (int j = 0; j < 4; ++j) acc[j] = (v8f){};

    // Compute one K chunk from LDS buffer `bufp` with A chunk pointer `apc`.
    auto compute_chunk = [&](const __bf16* bufp, const __bf16* apc) {
#pragma unroll 2
        for (int k0 = 0; k0 < KCHUNK; k0 += 32) {
            v8bf alo = *(const v8bf*)(apc + k0);
            v8bf ahi = *(const v8bf*)(apc + k0 + 16);
            v16bf afrag = __builtin_shufflevector(alo, ahi,
                0,1,2,3,4,5,6,7,8,9,10,11,12,13,14,15);
            const __bf16* bbase = bufp + k0 + half * 8;
            v16bf bfrag[4];
#pragma unroll
            for (int j = 0; j < 4; ++j) {            // all DS loads first
                const __bf16* bp = bbase + (size_t)(j * 16 + r) * KCHUNK;
                v8bf blo = *(const v8bf*)(bp);
                v8bf bhi = *(const v8bf*)(bp + 16);
                bfrag[j] = __builtin_shufflevector(blo, bhi,
                    0,1,2,3,4,5,6,7,8,9,10,11,12,13,14,15);
            }
#pragma unroll
            for (int j = 0; j < 4; ++j)              // WMMAs back-to-back
                acc[j] = __builtin_amdgcn_wmma_f32_16x16x32_bf16(
                    false, afrag, false, bfrag[j], (short)0, acc[j], false, false);
        }
    };

#if HAS_TDM
    // Issue TDM load of W[n0..n0+63][kc0..kc0+KCHUNK) into LDS buffer `bi`.
    auto issue_panel = [&](int bi, int kc0) {
        unsigned long long ga =
            (unsigned long long)reinterpret_cast<uintptr_t>(W + (size_t)n0 * KT + kc0);
        unsigned ldsOff = (unsigned)reinterpret_cast<uintptr_t>(&ldsW[bi][0]);
        v4u g0;
        g0[0] = 1u;                                  // count = 1 valid descriptor
        g0[1] = ldsOff;                              // lds_addr (bytes)
        g0[2] = (unsigned)ga;                        // global_addr[31:0]
        g0[3] = (unsigned)((ga >> 32) & 0x1FFFFFFu) | (2u << 30); // addr[56:32]|type=2
        v8i g1;
        g1[0] = (1 << 16);                           // data_size = 1 (2 bytes)
        g1[1] = (KCHUNK & 0xFFFF) << 16;             // tensor_dim0[15:0]
        g1[2] = ((KCHUNK >> 16) & 0xFFFF) | (64 << 16);  // dim0[31:16] | dim1[15:0]
        g1[3] = (KCHUNK & 0xFFFF) << 16;             // dim1[31:16]=0 | tile_dim0
        g1[4] = 64;                                  // tile_dim1 = 64, tile_dim2 = 0
        g1[5] = KT;                                  // tensor_dim0_stride (= ldw)
        g1[6] = 0;
        g1[7] = 0;
        v4i gz4 = {0, 0, 0, 0};
        v8i gz8 = {0, 0, 0, 0, 0, 0, 0, 0};
        __builtin_amdgcn_tensor_load_to_lds(g0, g1, gz4, gz4, gz8, 0);
    };

    // Prologue: stage chunk 0 into buffer 0.
    if (threadIdx.x < 32) {
        issue_panel(0, 0);
        __builtin_amdgcn_s_wait_tensorcnt(0);
    }
    __syncthreads();

    int buf = 0;
#pragma unroll
    for (int c = 0; c < NCHUNKS; ++c) {
        const int kc0 = c * KCHUNK;
        if (c + 1 < NCHUNKS && threadIdx.x < 32)
            issue_panel(buf ^ 1, kc0 + KCHUNK);      // async prefetch of next panel
        compute_chunk(&ldsW[buf][0], a_chunk_ptr(kc0));
        if (c + 1 < NCHUNKS) {
            if (threadIdx.x < 32) __builtin_amdgcn_s_wait_tensorcnt(0);
            __syncthreads();
            buf ^= 1;
        }
    }
#else
    // Fallback: synchronous cooperative staging, single buffer.
    for (int c = 0; c < NCHUNKS; ++c) {
        const int kc0 = c * KCHUNK;
        if (c > 0) __syncthreads();
        for (int idx = threadIdx.x; idx < 64 * (KCHUNK / 8); idx += 256) {
            int row = idx / (KCHUNK / 8);
            int cc  = (idx % (KCHUNK / 8)) * 8;
            *(v8bf*)(&ldsW[0][0] + row * KCHUNK + cc) =
                *(const v8bf*)(W + (size_t)(n0 + row) * KT + kc0 + cc);
        }
        __syncthreads();
        compute_chunk(&ldsW[0][0], a_chunk_ptr(kc0));
    }
#endif

    // ---- epilogue ----
#pragma unroll
    for (int j = 0; j < 4; ++j) {
        const int col = n0 + j * 16 + r;
        const float bv = SOFTPLUS ? bias[col] : 0.f;
#pragma unroll
        for (int i = 0; i < 8; ++i) {
            const int row = tm + i + half * 8;
            float v = acc[j][i];
            if (SOFTPLUS) v = softplusf(v + bv);
            if (WF32) outf[(size_t)row * ldo + col] = v;
            if (WBF)  outb[(size_t)row * ldo + col] = f2bf(v);
        }
    }
    (void)M;
}

// Depthwise conv (k=4, pad left=1/right=2) + SiLU; writes f32 and bf16 copies.
__global__ void __launch_bounds__(256)
conv_silu_kernel(const __bf16* __restrict__ xz,
                 const float* __restrict__ wx, const float* __restrict__ bx,
                 const float* __restrict__ wz, const float* __restrict__ bz,
                 float* __restrict__ outf, __bf16* __restrict__ outb) {
    const int t = blockIdx.x * blockDim.x + threadIdx.x;   // 0 .. 8191
    const int b = t >> 10;
    const int d = t & (D_INNER - 1);
    const float* w; float bias;
    if (d < D_HALF) { w = wx + d * 4;            bias = bx[d]; }
    else            { w = wz + (d - D_HALF) * 4; bias = bz[d - D_HALF]; }
    const float w0 = w[0], w1 = w[1], w2 = w[2], w3 = w[3];

    const __bf16* in = xz + (size_t)b * SEQ * D_INNER + d;
    float*   opf = outf + (size_t)b * SEQ * D_INNER + d;
    __bf16*  opb = outb + (size_t)b * SEQ * D_INNER + d;

    float um1 = 0.f;
    float u0  = bf2f(in[0]);
    float up1 = bf2f(in[D_INNER]);
    for (int l = 0; l < SEQ; ++l) {
        __builtin_prefetch(in + (size_t)(l + 16) * D_INNER, 0, 0);
        float up2 = (l + 2 < SEQ) ? bf2f(in[(size_t)(l + 2) * D_INNER]) : 0.f;
        float a = w0 * um1 + w1 * u0 + w2 * up1 + w3 * up2 + bias;
        float s = a / (1.f + __expf(-a));                  // SiLU
        opf[(size_t)l * D_INNER] = s;
        opb[(size_t)l * D_INNER] = f2bf(s);
        um1 = u0; u0 = up1; up1 = up2;
    }
}

// Selective scan: one thread per (b, d); h[16] in VGPRs; writes y as bf16.
__global__ void __launch_bounds__(256)
scan_kernel(const float* __restrict__ conv_out,   // [b, l, 1024]; u = col d
            const float* __restrict__ dtb,        // [b, l, 512]
            const float* __restrict__ xdbl,       // [b, l, 64]; B=32..47, C=48..63
            const float* __restrict__ A_log,      // [512, 16]
            const float* __restrict__ Dp,         // [512]
            __bf16* __restrict__ y) {             // [b, l, 512] bf16
    const int t = blockIdx.x * blockDim.x + threadIdx.x;   // 0 .. 4095
    const int b = t >> 9;
    const int d = t & (D_HALF - 1);

    float Av[N_STATE];
#pragma unroll
    for (int n = 0; n < N_STATE; ++n) Av[n] = -__expf(A_log[d * N_STATE + n]);
    const float Dv = Dp[d];

    float h[N_STATE];
#pragma unroll
    for (int n = 0; n < N_STATE; ++n) h[n] = 0.f;

    const float* up  = conv_out + (size_t)b * SEQ * D_INNER + d;
    const float* dtp = dtb      + (size_t)b * SEQ * D_HALF  + d;
    const float* bcp = xdbl     + (size_t)b * SEQ * 64 + DT_RANK;
    __bf16*      yp  = y        + (size_t)b * SEQ * D_HALF  + d;

    for (int l = 0; l < SEQ; ++l) {
        __builtin_prefetch(up  + (size_t)(l + 8) * D_INNER, 0, 0);
        __builtin_prefetch(dtp + (size_t)(l + 8) * D_HALF, 0, 0);
        const float u   = up[(size_t)l * D_INNER];
        const float dtv = dtp[(size_t)l * D_HALF];
        const float4* bc = (const float4*)(bcp + (size_t)l * 64);
        float4 B0 = bc[0], B1 = bc[1], B2 = bc[2], B3 = bc[3];
        float4 C0 = bc[4], C1 = bc[5], C2 = bc[6], C3 = bc[7];
        const float Bv[N_STATE] = {B0.x,B0.y,B0.z,B0.w, B1.x,B1.y,B1.z,B1.w,
                                   B2.x,B2.y,B2.z,B2.w, B3.x,B3.y,B3.z,B3.w};
        const float Cv[N_STATE] = {C0.x,C0.y,C0.z,C0.w, C1.x,C1.y,C1.z,C1.w,
                                   C2.x,C2.y,C2.z,C2.w, C3.x,C3.y,C3.z,C3.w};
        const float du = dtv * u;
        float acc = 0.f;
#pragma unroll
        for (int n = 0; n < N_STATE; ++n) {
            const float dA = __expf(dtv * Av[n]);
            h[n] = h[n] * dA + du * Bv[n];
            acc  = fmaf(h[n], Cv[n], acc);
        }
        yp[(size_t)l * D_HALF] = f2bf(acc + Dv * u);
    }
}

extern "C" void kernel_launch(void* const* d_in, const int* in_sizes, int n_in,
                              void* d_out, int out_size, void* d_ws, size_t ws_size,
                              hipStream_t stream) {
    (void)in_sizes; (void)n_in; (void)out_size; (void)ws_size;
    const float* hidden    = (const float*)d_in[0];   // [8,2048,512]
    const float* in_proj_w = (const float*)d_in[1];   // [1024,512]
    const float* conv_x_w  = (const float*)d_in[2];
    const float* conv_x_b  = (const float*)d_in[3];
    const float* conv_z_w  = (const float*)d_in[4];
    const float* conv_z_b  = (const float*)d_in[5];
    const float* x_proj_w  = (const float*)d_in[6];   // [64,512]
    const float* dt_proj_w = (const float*)d_in[7];   // [512,32]
    const float* dt_proj_b = (const float*)d_in[8];
    const float* A_log     = (const float*)d_in[9];
    const float* Dp        = (const float*)d_in[10];
    const float* out_proj_w= (const float*)d_in[11];  // [512,1024]
    float* out = (float*)d_out;                       // [8,2048,512]

    // ---- workspace carve-up (byte offsets; regions aliased per liveness) ----
    char* p = (char*)d_ws;
    float*  conv_out  = (float*)(p);                       // 64 MB  (persists)
    char*   regionA   = p + (64u << 20);                   // 32 MB  xz_bf -> dt f32
    __bf16* xz_bf     = (__bf16*)regionA;
    float*  dtbuf     = (float*)regionA;                   // alias (xz dead after conv)
    __bf16* conv_bf   = (__bf16*)(p + (96u << 20));        // 32 MB  (persists)
    char*   regionB   = p + (128u << 20);                  // 16 MB  hidden_bf -> y_bf
    __bf16* hidden_bf = (__bf16*)regionB;
    __bf16* y_bf      = (__bf16*)regionB;                  // alias (hidden_bf dead)
    float*  x_dbl     = (float*)(p + (144u << 20));        // 4 MB
    __bf16* x_dbl_bf  = (__bf16*)(p + (148u << 20));       // 2 MB
    __bf16* w1_bf     = (__bf16*)(p + (150u << 20));       // in_proj  1 MB
    __bf16* w2_bf     = (__bf16*)(p + (152u << 20));       // x_proj   64 KB
    __bf16* w3_bf     = (__bf16*)(p + (153u << 20));       // dt_proj  32 KB
    __bf16* w4_bf     = (__bf16*)(p + (154u << 20));       // out_proj 1 MB

    const dim3 blk(256);
    auto castN = [&](const float* src, __bf16* dst, int n) {
        cast_f32_bf16<<<(n / 4 + 255) / 256, blk, 0, stream>>>(src, dst, n);
    };
    auto nblocks = [](int M, int N) { return (M >> 7) * (N >> 6); };

    // 0) one-off bf16 casts
    castN(hidden,     hidden_bf, M_ROWS * D_MODEL);
    castN(in_proj_w,  w1_bf, D_INNER * D_MODEL);
    castN(x_proj_w,   w2_bf, (DT_RANK + 2 * N_STATE) * D_HALF);
    castN(dt_proj_w,  w3_bf, D_HALF * DT_RANK);
    castN(out_proj_w, w4_bf, D_MODEL * D_INNER);

    // 1) in_proj: xz_bf[16384,1024] = hidden @ in_proj_w^T
    gemm_wmma_panel<D_MODEL, false, false, true>
        <<<nblocks(M_ROWS, D_INNER), blk, 0, stream>>>(
        hidden_bf, D_MODEL, hidden_bf, D_MODEL, D_MODEL,
        w1_bf, nullptr, nullptr, xz_bf, D_INNER, M_ROWS, D_INNER);

    // 2) depthwise conv + SiLU -> conv_out (f32) + conv_bf (bf16)
    conv_silu_kernel<<<(B_SZ * D_INNER) / 256, blk, 0, stream>>>(
        xz_bf, conv_x_w, conv_x_b, conv_z_w, conv_z_b, conv_out, conv_bf);

    // 3) x_proj: x_dbl[16384,64] (f32 + bf16) = x_conv @ x_proj_w^T
    gemm_wmma_panel<D_HALF, false, true, true>
        <<<nblocks(M_ROWS, DT_RANK + 2 * N_STATE), blk, 0, stream>>>(
        conv_bf, D_INNER, conv_bf, D_INNER, D_HALF,
        w2_bf, nullptr, x_dbl, x_dbl_bf, DT_RANK + 2 * N_STATE,
        M_ROWS, DT_RANK + 2 * N_STATE);

    // 4) dt_proj + bias + softplus -> dtbuf[16384,512] f32 (K = 32)
    gemm_wmma_panel<DT_RANK, true, true, false>
        <<<nblocks(M_ROWS, D_HALF), blk, 0, stream>>>(
        x_dbl_bf, DT_RANK + 2 * N_STATE, x_dbl_bf, DT_RANK + 2 * N_STATE, DT_RANK,
        w3_bf, dt_proj_b, dtbuf, nullptr, D_HALF, M_ROWS, D_HALF);

    // 5) selective scan -> y_bf[16384,512]
    scan_kernel<<<(B_SZ * D_HALF) / 256, blk, 0, stream>>>(
        conv_out, dtbuf, x_dbl, A_log, Dp, y_bf);

    // 6) out_proj: out = [y | z_conv] @ out_proj_w^T  (split-A, K = 1024)
    gemm_wmma_panel<D_INNER, false, true, false>
        <<<nblocks(M_ROWS, D_MODEL), blk, 0, stream>>>(
        y_bf, D_HALF, conv_bf + D_HALF, D_INNER, D_HALF,
        w4_bf, nullptr, out, nullptr, D_MODEL, M_ROWS, D_MODEL);
}
